// MultiScaleVQVAE_28750511079587
// MI455X (gfx1250) — matrix operand order repaired and checked by reading, not compiled, not run on hardware
//
#include <hip/hip_runtime.h>
#include <cstdint>
#include <cstddef>

// ---------------------------------------------------------------------------
// MultiScaleVQVAE forward for MI455X (gfx1250, wave32, WMMA).
//  * one-hot first conv done as gather (one-hot conv == weight LUT)
//  * all dense 128-channel convs as implicit-GEMM with v_wmma_f32_16x16x32_f16
//    - A tile gathered with swizzle-on-store: fragment = contiguous 32B LDS read
//    - B weights pre-packed in fragment order; staged via async global->LDS b128
//    - 2 WMMAs per wave per K-step (16Mx32N per wave, 16Mx128N per workgroup)
//  * VQ argmin / resizes / convT / BCE as VALU kernels with LDS reductions
// ---------------------------------------------------------------------------

typedef __attribute__((ext_vector_type(16))) _Float16 v16h;
typedef __attribute__((ext_vector_type(8)))  float    v8f;
typedef __attribute__((ext_vector_type(4)))  int      v4i_;

#if defined(__gfx1250__) && __has_builtin(__builtin_amdgcn_global_load_async_to_lds_b128) && __has_builtin(__builtin_amdgcn_s_wait_asynccnt)
#define HAVE_ASYNC_LDS 1
#else
#define HAVE_ASYNC_LDS 0
#endif

#define BN   2
#define ZC   16
#define NCLS 18
#define YD   200
#define XD   200
#define C0   64        // Hc/2
#define CD   128       // Hc == D
#define KC   4096      // codebook entries
#define HLAT 25

static const int NLOGITS = BN * ZC * NCLS * YD * XD;   // 23,040,000

// ---------------------------------------------------------------------------
// small helpers
// ---------------------------------------------------------------------------
__global__ void zero_f_kernel(float* p, int n) {
  int i = blockIdx.x * blockDim.x + threadIdx.x;
  if (i < n) p[i] = 0.f;
}
__global__ void zero_i_kernel(int* p, int n) {
  int i = blockIdx.x * blockDim.x + threadIdx.x;
  if (i < n) p[i] = 0;
}
__global__ void f32_to_f16_kernel(const float* __restrict__ a, _Float16* __restrict__ b, int n) {
  int i = blockIdx.x * blockDim.x + threadIdx.x;
  if (i < n) b[i] = (_Float16)a[i];
}

// ---------------------------------------------------------------------------
// pack conv weight (Cout, Cin, kh, kw) fp32 -> fragment-ordered fp16:
//   dst[kblock][nblock][lane][e]  with
//     kblock = k/32, kk = k%32, nblock = co/16, nl = co%16
//     lane = nl + 16*(kk>=16)        (contiguous k order, same as A staging: lanes 0-15 K 0..15,
//     e    = kk & 15                  lanes 16-31 K 16..31; VGPR j = K pair)
// so one lane's v16h B-fragment is 32 contiguous bytes, and one fragment
// block (512 halfs) is contiguous for coalesced async b128 copies.
// ---------------------------------------------------------------------------
__global__ void pack_w_kernel(const float* __restrict__ w, _Float16* __restrict__ wpk,
                              int Cout, int Cin, int khw) {
  int i = blockIdx.x * blockDim.x + threadIdx.x;
  int n = Cout * Cin * khw;
  if (i >= n) return;
  int co = i / (Cin * khw);
  int k  = i - co * (Cin * khw);
  int kblock = k >> 5, kk = k & 31;
  int nblock = co >> 4, nl = co & 15;
  int lane = nl + ((kk >> 4) << 4);
  int e    = kk & 15;
  size_t dst = (((size_t)kblock * (Cout >> 4) + nblock) * 32 + lane) * 16 + e;
  wpk[dst] = (_Float16)w[i];
}

// ---------------------------------------------------------------------------
// first conv over one-hot input: pure gather-accumulate (no multiplies needed)
// ---------------------------------------------------------------------------
__global__ void __launch_bounds__(64)
comp_onehot_kernel(const int* __restrict__ x, const float* __restrict__ w,
                   float* __restrict__ out) {
  int pix = blockIdx.x;                       // b*Y*X + y*X + x
  int b  = pix / (YD * XD);
  int r  = pix - b * (YD * XD);
  int yy = r / XD;
  int xx = r - yy * XD;

  __shared__ int cls[ZC * 9];
  for (int t = threadIdx.x; t < ZC * 9; t += 64) {
    int z = t / 9, tap = t - z * 9;
    int dy = tap / 3 - 1, dx = tap % 3 - 1;
    int sy = yy + dy, sx = xx + dx;
    cls[t] = (sy >= 0 && sy < YD && sx >= 0 && sx < XD)
                 ? x[((size_t)(b * ZC + z) * YD + sy) * XD + sx] : -1;
  }
  __syncthreads();

  int co = threadIdx.x;                        // 64 output channels
  float acc = 0.f;
  for (int t = 0; t < ZC * 9; ++t) {
    int c = cls[t];
    if (c >= 0) {
      int z = t / 9, tap = t - z * 9;
      acc += w[((size_t)co * (ZC * NCLS) + z * NCLS + c) * 9 + tap];
    }
  }
  out[((size_t)(b * C0 + co) * YD + yy) * XD + xx] = acc;
}

// batch-norm statistics (training mode, biased variance), one block per channel
__global__ void __launch_bounds__(256)
bn_stats_kernel(const float* __restrict__ h, float* __restrict__ mean,
                float* __restrict__ rstd, int HW) {
  int c = blockIdx.x;
  float s = 0.f, ss = 0.f;
  for (int b = 0; b < BN; ++b) {
    const float* p = h + ((size_t)(b * C0 + c)) * HW;
    for (int i = threadIdx.x; i < HW; i += 256) {
      float v = p[i];
      s += v; ss += v * v;
    }
  }
  __shared__ float sh1[256], sh2[256];
  sh1[threadIdx.x] = s; sh2[threadIdx.x] = ss;
  __syncthreads();
  for (int t = 128; t > 0; t >>= 1) {
    if (threadIdx.x < t) { sh1[threadIdx.x] += sh1[threadIdx.x + t]; sh2[threadIdx.x] += sh2[threadIdx.x + t]; }
    __syncthreads();
  }
  if (threadIdx.x == 0) {
    float N = (float)(BN * HW);
    float m = sh1[0] / N;
    float var = sh2[0] / N - m * m;
    mean[c] = m;
    rstd[c] = rsqrtf(var + 1e-5f);
  }
}

__global__ void bn_relu_f16_kernel(const float* __restrict__ h, const float* __restrict__ mean,
                                   const float* __restrict__ rstd, const float* __restrict__ g,
                                   const float* __restrict__ bb, _Float16* __restrict__ out,
                                   int HW, int n) {
  int i = blockIdx.x * blockDim.x + threadIdx.x;
  if (i >= n) return;
  int c = (i / HW) % C0;
  float v = (h[i] - mean[c]) * rstd[c] * g[c] + bb[c];
  out[i] = (_Float16)fmaxf(v, 0.f);
}

// ---------------------------------------------------------------------------
// implicit-GEMM conv with WMMA (wave32).
// M = B*Ho*Wo pixels, N = Cout = 128, K = Cin*kh*kw (K%32==0 for all calls).
// Workgroup = 128 threads = 4 waves; wave w covers N in [w*32, w*32+32).
// Per K-step: async-copy two fragment-packed B tiles to LDS, im2col-gather A
// with swizzle-on-store so fragments are contiguous 32B, then 2x WMMA.
// accum: 0 = store, +1 = out += v, -1 = out -= v  (v optionally relu'd)
// ---------------------------------------------------------------------------
__global__ void __launch_bounds__(128)
conv_wmma_kernel(const _Float16* __restrict__ in, const _Float16* __restrict__ wpk,
                 const float* __restrict__ bias, float* out32, _Float16* out16,
                 int Cin, int H, int W, int Cout, int Ho, int Wo,
                 int kh, int kw, int stride, int pad, int relu, int accum) {
  const int Ktot  = Cin * kh * kw;
  const int khw   = kh * kw;
  const int HoWo  = Ho * Wo;
  const int M     = BN * HoWo;
  const int tileM = blockIdx.x;
  const int tid   = threadIdx.x;
  const int wv    = tid >> 5;
  const int lane  = tid & 31;
  const int nblocks = Cout >> 4;               // fragment blocks along N

  // A: one 16x32 tile in fragment order -> lane-major [32 lanes][16 halfs]
  __shared__ __align__(32) _Float16 ldsA[32][16];
  // B: per wave, two 32x16 fragments in fragment order
  __shared__ __align__(32) _Float16 ldsB[4][2][512];

  v8f acc0 = {};
  v8f acc1 = {};

  for (int k0 = 0; k0 < Ktot; k0 += 32) {
    const int kblock = k0 >> 5;

    // ---- (1) issue async B-fragment copies first (overlap with A gather) ----
#if HAVE_ASYNC_LDS
    {
      typedef __attribute__((address_space(1))) v4i_ gv4i_t;
      typedef __attribute__((address_space(3))) v4i_ sv4i_t;
      #pragma unroll
      for (int j = 0; j < 2; ++j) {
        const _Float16* gsrc =
            wpk + (((size_t)kblock * nblocks + (wv * 2 + j)) << 9) + lane * 16;
        _Float16* ldst = &ldsB[wv][j][lane * 16];
        __builtin_amdgcn_global_load_async_to_lds_b128(
            (gv4i_t*)(v4i_*)(void*)gsrc, (sv4i_t*)(v4i_*)(void*)ldst, 0, 0);
        __builtin_amdgcn_global_load_async_to_lds_b128(
            (gv4i_t*)(v4i_*)(void*)(gsrc + 8), (sv4i_t*)(v4i_*)(void*)(ldst + 8), 0, 0);
      }
    }
#else
    {
      #pragma unroll
      for (int j = 0; j < 2; ++j) {
        const _Float16* gsrc =
            wpk + (((size_t)kblock * nblocks + (wv * 2 + j)) << 9) + lane * 16;
        *(v16h*)&ldsB[wv][j][lane * 16] = *(const v16h*)gsrc;
      }
    }
#endif
    if (lane == 0 && k0 + 32 < Ktot)
      __builtin_prefetch(wpk + (((size_t)(kblock + 1) * nblocks) << 9), 0, 1);

    // ---- (2) stage A tile with swizzle-on-store (fragment order) ----
    #pragma unroll
    for (int j = 0; j < 4; ++j) {
      int e  = tid * 4 + j;             // 0..511
      int m  = e >> 5;
      int kk = e & 31;
      _Float16 val = (_Float16)0.0f;
      int mg = tileM * 16 + m;
      int k  = k0 + kk;
      if (mg < M && k < Ktot) {
        int cin = k / khw;
        int t   = k - cin * khw;
        int dy  = t / kw;
        int dx  = t - dy * kw;
        int b   = mg / HoWo;
        int rr  = mg - b * HoWo;
        int oy  = rr / Wo;
        int ox  = rr - oy * Wo;
        int iy  = oy * stride - pad + dy;
        int ix  = ox * stride - pad + dx;
        if (iy >= 0 && iy < H && ix >= 0 && ix < W)
          val = in[((size_t)(b * Cin + cin) * H + iy) * W + ix];
      }
      // A fragment order used here (contiguous k, as in pack_w): lane = m + 16*(kk>=16),
      // element = kk&15
      int alane = m + ((kk >> 4) << 4);
      int ae    = kk & 15;
      ldsA[alane][ae] = val;
    }

#if HAVE_ASYNC_LDS
    __builtin_amdgcn_s_wait_asynccnt(0);
#endif
    __syncthreads();

    // ---- (3) contiguous fragment loads + 2x WMMA ----
    v16h afrag = *(const v16h*)&ldsA[lane][0];
    v16h bfrag0 = *(const v16h*)&ldsB[wv][0][lane * 16];
    v16h bfrag1 = *(const v16h*)&ldsB[wv][1][lane * 16];
    acc0 = __builtin_amdgcn_wmma_f32_16x16x32_f16(false, afrag, false, bfrag0,
                                                  (short)0, acc0, false, false);
    acc1 = __builtin_amdgcn_wmma_f32_16x16x32_f16(false, afrag, false, bfrag1,
                                                  (short)0, acc1, false, false);
    __syncthreads();
  }

  // ---- write back: C/D 16x16 f32 layout: N=lane&15; VGPR r -> M=r(+8 hi lanes)
  const int m0 = tileM * 16 + ((lane >> 4) ? 8 : 0);
  #pragma unroll
  for (int j = 0; j < 2; ++j) {
    const int n  = wv * 32 + j * 16 + (lane & 15);
    const float bv = bias ? bias[n] : 0.f;
    #pragma unroll
    for (int r = 0; r < 8; ++r) {
      int mg = m0 + r;
      if (mg >= M) continue;
      float v = (j == 0 ? acc0[r] : acc1[r]) + bv;
      if (relu) v = fmaxf(v, 0.f);
      int b  = mg / HoWo;
      int rr = mg - b * HoWo;
      int oy = rr / Wo;
      int ox = rr - oy * Wo;
      size_t oi = ((size_t)(b * Cout + n) * Ho + oy) * Wo + ox;
      if (out32) {
        if (accum == 0)      out32[oi]  = v;
        else if (accum > 0)  out32[oi] += v;
        else                 out32[oi] -= v;
      }
      if (out16) out16[oi] = (_Float16)v;
    }
  }
}

// ---------------------------------------------------------------------------
// VQ: codebook row norms, per-position argmin + vq-loss, perplexity
// ---------------------------------------------------------------------------
__global__ void cnorm_kernel(const float* __restrict__ cbk, float* __restrict__ cn) {
  int c = blockIdx.x * blockDim.x + threadIdx.x;
  if (c >= KC) return;
  const float* r = cbk + (size_t)c * CD;
  float s = 0.f;
  for (int d = 0; d < CD; ++d) s += r[d] * r[d];
  cn[c] = s;
}

__global__ void __launch_bounds__(256)
vq_kernel(const float* __restrict__ fhat, const float* __restrict__ cbk,
          const float* __restrict__ cn, int* __restrict__ idx, int* __restrict__ counts,
          float* __restrict__ vq_accum, int S, float loss_scale) {
  int p = blockIdx.x;                   // position in [0, BN*S*S)
  int b = p / (S * S);
  int r = p - b * (S * S);
  int i = r / S, j = r - (r / S) * S;
  int tid = threadIdx.x;

  __shared__ float z[CD];
  if (tid < CD) z[tid] = fhat[((size_t)(b * CD + tid) * S + i) * S + j];
  __syncthreads();

  float bd = 3.4e38f; int bi = 0;
  for (int c = tid; c < KC; c += 256) {
    const float* cr = cbk + (size_t)c * CD;
    float dot = 0.f;
    #pragma unroll 4
    for (int d = 0; d < CD; ++d) dot += z[d] * cr[d];
    float dist = cn[c] - 2.f * dot;
    if (dist < bd || (dist == bd && c < bi)) { bd = dist; bi = c; }
  }
  __shared__ float bval[256];
  __shared__ int   bidx[256];
  bval[tid] = bd; bidx[tid] = bi;
  __syncthreads();
  for (int t = 128; t > 0; t >>= 1) {
    if (tid < t) {
      if (bval[tid + t] < bval[tid] ||
          (bval[tid + t] == bval[tid] && bidx[tid + t] < bidx[tid])) {
        bval[tid] = bval[tid + t]; bidx[tid] = bidx[tid + t];
      }
    }
    __syncthreads();
  }
  int best = bidx[0];
  if (tid == 0) { idx[p] = best; atomicAdd(&counts[best], 1); }

  __shared__ float dsum[CD];
  if (tid < CD) {
    float d = cbk[(size_t)best * CD + tid] - z[tid];
    dsum[tid] = d * d;
  }
  __syncthreads();
  for (int t = 64; t > 0; t >>= 1) {
    if (tid < t) dsum[tid] += dsum[tid + t];
    __syncthreads();
  }
  if (tid == 0) atomicAdd(vq_accum, dsum[0] * loss_scale);
}

__global__ void __launch_bounds__(256)
perp_kernel(const int* __restrict__ counts, float* __restrict__ out, float invP) {
  __shared__ float sh[256];
  float s = 0.f;
  for (int c = threadIdx.x; c < KC; c += 256) {
    float pr = counts[c] * invP;
    s += pr * logf(pr + 1e-10f);
  }
  sh[threadIdx.x] = s;
  __syncthreads();
  for (int t = 128; t > 0; t >>= 1) {
    if (threadIdx.x < t) sh[threadIdx.x] += sh[threadIdx.x + t];
    __syncthreads();
  }
  if (threadIdx.x == 0) *out = expf(-sh[0]);
}

// ---------------------------------------------------------------------------
// area downsample 25 -> S (adaptive average pooling)
// ---------------------------------------------------------------------------
__global__ void area_kernel(const float* __restrict__ f, float* __restrict__ fhat, int S) {
  int i = blockIdx.x * blockDim.x + threadIdx.x;
  int tot = BN * CD * S * S;
  if (i >= tot) return;
  int jx = i % S; int t = i / S;
  int iy = t % S; t /= S;
  int c  = t % CD; int b = t / CD;
  int r0 = (iy * HLAT) / S, r1 = ((iy + 1) * HLAT + S - 1) / S;
  int c0 = (jx * HLAT) / S, c1 = ((jx + 1) * HLAT + S - 1) / S;
  float s = 0.f;
  for (int y = r0; y < r1; ++y)
    for (int xx = c0; xx < c1; ++xx)
      s += f[((size_t)(b * CD + c) * HLAT + y) * HLAT + xx];
  fhat[i] = s / (float)((r1 - r0) * (c1 - c0));
}

// ---------------------------------------------------------------------------
// bicubic upsample of codebook[idx] (S -> 25), a = -0.75, border clamp; fp16 out
// ---------------------------------------------------------------------------
__device__ inline float cubicw(float xv) {
  const float a = -0.75f;
  float x = fabsf(xv);
  if (x <= 1.f) return ((a + 2.f) * x - (a + 3.f)) * x * x + 1.f;
  if (x < 2.f)  return (((x - 5.f) * x + 8.f) * x - 4.f) * a;
  return 0.f;
}

__global__ void bicubic_z_kernel(const int* __restrict__ idx, const float* __restrict__ cbk,
                                 _Float16* __restrict__ z16, int S) {
  int i = blockIdx.x * blockDim.x + threadIdx.x;
  int tot = BN * CD * HLAT * HLAT;
  if (i >= tot) return;
  int px = i % HLAT; int t = i / HLAT;
  int py = t % HLAT; t /= HLAT;
  int c  = t % CD; int b = t / CD;
  float scale = (float)S / (float)HLAT;
  float ys = (py + 0.5f) * scale - 0.5f;
  float xs = (px + 0.5f) * scale - 0.5f;
  int iy0 = (int)floorf(ys); float ty = ys - iy0;
  int ix0 = (int)floorf(xs); float tx = xs - ix0;
  float wy[4] = {cubicw(ty + 1.f), cubicw(ty), cubicw(1.f - ty), cubicw(2.f - ty)};
  float wx[4] = {cubicw(tx + 1.f), cubicw(tx), cubicw(1.f - tx), cubicw(2.f - tx)};
  float acc = 0.f;
  for (int ky = 0; ky < 4; ++ky) {
    int ry = iy0 - 1 + ky; ry = ry < 0 ? 0 : (ry >= S ? S - 1 : ry);
    for (int kx = 0; kx < 4; ++kx) {
      int rx = ix0 - 1 + kx; rx = rx < 0 ? 0 : (rx >= S ? S - 1 : rx);
      int code = idx[(size_t)(b * S + ry) * S + rx];
      acc += wy[ky] * wx[kx] * cbk[(size_t)code * CD + c];
    }
  }
  z16[i] = (_Float16)acc;
}

// ---------------------------------------------------------------------------
// ConvTranspose2d (k=4, stride=2, pad=1), weight layout (Cin, Cout, 4, 4)
// ---------------------------------------------------------------------------
__global__ void convT_kernel(const float* __restrict__ in, const float* __restrict__ w,
                             const float* __restrict__ bias, float* __restrict__ out,
                             int Cin, int Hin, int Win, int Cout, int Ho, int Wo, int relu) {
  size_t i = (size_t)blockIdx.x * blockDim.x + threadIdx.x;
  size_t tot = (size_t)BN * Cout * Ho * Wo;
  if (i >= tot) return;
  int ox = i % Wo; size_t t = i / Wo;
  int oy = t % Ho; t /= Ho;
  int co = t % Cout; int b = t / Cout;

  int kys[2], iys[2], nky = 0;
  for (int ky = 0; ky < 4; ++ky) {
    int num = oy + 1 - ky;
    if (num >= 0 && (num & 1) == 0) {
      int iy = num >> 1;
      if (iy < Hin) { kys[nky] = ky; iys[nky] = iy; ++nky; }
    }
  }
  int kxs[2], ixs[2], nkx = 0;
  for (int kx = 0; kx < 4; ++kx) {
    int num = ox + 1 - kx;
    if (num >= 0 && (num & 1) == 0) {
      int ix = num >> 1;
      if (ix < Win) { kxs[nkx] = kx; ixs[nkx] = ix; ++nkx; }
    }
  }
  float acc = bias[co];
  for (int ci = 0; ci < Cin; ++ci) {
    const float* wr = w + ((size_t)ci * Cout + co) * 16;
    const float* ir = in + ((size_t)(b * Cin + ci) * Hin) * Win;
    for (int a = 0; a < nky; ++a)
      for (int bb = 0; bb < nkx; ++bb)
        acc += wr[kys[a] * 4 + kxs[bb]] * ir[(size_t)iys[a] * Win + ixs[bb]];
  }
  if (relu) acc = fmaxf(acc, 0.f);
  out[i] = acc;
}

// 1x1 decomp conv 64 -> 288, writes logits into d_out
__global__ void decomp_kernel(const float* __restrict__ h, const float* __restrict__ w,
                              const float* __restrict__ bias, float* __restrict__ out) {
  size_t i = (size_t)blockIdx.x * blockDim.x + threadIdx.x;
  const size_t tot = (size_t)BN * ZC * NCLS * YD * XD;
  if (i >= tot) return;
  const int HW = YD * XD;
  int p = i % HW; size_t t = i / HW;
  int co = t % (ZC * NCLS); int b = t / (ZC * NCLS);
  float acc = bias[co];
  const float* hb = h + (size_t)b * C0 * HW + p;
  const float* wr = w + (size_t)co * C0;
  for (int ci = 0; ci < C0; ++ci) acc += wr[ci] * hb[(size_t)ci * HW];
  out[i] = acc;
}

// BCE-with-logits vs one-hot(x), mean reduction into a scalar
__global__ void __launch_bounds__(256)
bce_kernel(const float* __restrict__ logits, const int* __restrict__ x,
           float* __restrict__ accum) {
  const size_t N = (size_t)BN * ZC * NCLS * YD * XD;
  const int YX = YD * XD;
  float s = 0.f;
  for (size_t i = (size_t)blockIdx.x * 256 + threadIdx.x; i < N; i += (size_t)gridDim.x * 256) {
    int px = i % YX; size_t t = i / YX;
    int ch = t % (ZC * NCLS); int b = t / (ZC * NCLS);
    int z = ch / NCLS, cls = ch % NCLS;
    float tt = (x[(size_t)(b * ZC + z) * YX + px] == cls) ? 1.f : 0.f;
    float l = logits[i];
    s += fmaxf(l, 0.f) - l * tt + log1pf(expf(-fabsf(l)));
  }
  __shared__ float sh[256];
  sh[threadIdx.x] = s;
  __syncthreads();
  for (int t = 128; t > 0; t >>= 1) {
    if (threadIdx.x < t) sh[threadIdx.x] += sh[threadIdx.x + t];
    __syncthreads();
  }
  if (threadIdx.x == 0) atomicAdd(accum, sh[0] * (float)(1.0 / (double)N));
}

// ---------------------------------------------------------------------------
// host orchestration
// ---------------------------------------------------------------------------
extern "C" void kernel_launch(void* const* d_in, const int* in_sizes, int n_in,
                              void* d_out, int out_size, void* d_ws, size_t ws_size,
                              hipStream_t stream) {
  (void)in_sizes; (void)n_in; (void)out_size; (void)ws_size;
  const int*   x       = (const int*)  d_in[0];
  const float* cbk     = (const float*)d_in[1];
  const float* comp_w  = (const float*)d_in[2];
  const float* bn_g    = (const float*)d_in[3];
  const float* bn_bb   = (const float*)d_in[4];
  const float* enc_w1  = (const float*)d_in[5];
  const float* enc_b1  = (const float*)d_in[6];
  const float* enc_w2  = (const float*)d_in[7];
  const float* enc_b2  = (const float*)d_in[8];
  const float* enc_w3  = (const float*)d_in[9];
  const float* enc_b3  = (const float*)d_in[10];
  const float* enc_w4  = (const float*)d_in[11];
  const float* enc_b4  = (const float*)d_in[12];
  const float* phi_e_w = (const float*)d_in[13];
  const float* phi_e_b = (const float*)d_in[14];
  const float* phi_d_w = (const float*)d_in[15];
  const float* phi_d_b = (const float*)d_in[16];
  const float* dec_w0  = (const float*)d_in[17];
  const float* dec_b0  = (const float*)d_in[18];
  const float* dt_w1   = (const float*)d_in[19];
  const float* dt_b1   = (const float*)d_in[20];
  const float* dt_w2   = (const float*)d_in[21];
  const float* dt_b2   = (const float*)d_in[22];
  const float* dt_w3   = (const float*)d_in[23];
  const float* dt_b3   = (const float*)d_in[24];
  const float* dcmp_w  = (const float*)d_in[25];
  const float* dcmp_b  = (const float*)d_in[26];
  float* out = (float*)d_out;

  // workspace layout (256B aligned for b128 async loads)
  char* ws = (char*)d_ws;
  size_t off = 0;
  auto alloc = [&](size_t bytes) -> char* {
    char* p = ws + off;
    off = (off + bytes + 255) & ~(size_t)255;
    return p;
  };
  float*    h0_32  = (float*)   alloc((size_t)BN * C0 * YD * XD * 4);
  _Float16* h0_16  = (_Float16*)alloc((size_t)BN * C0 * YD * XD * 2);
  float*    bn_m   = (float*)   alloc(C0 * 4);
  float*    bn_r   = (float*)   alloc(C0 * 4);
  _Float16* e1_16  = (_Float16*)alloc((size_t)BN * CD * 100 * 100 * 2);
  _Float16* e2_16  = (_Float16*)alloc((size_t)BN * CD * 50 * 50 * 2);
  _Float16* e3_16  = (_Float16*)alloc((size_t)BN * CD * 25 * 25 * 2);
  float*    f32    = (float*)   alloc((size_t)BN * CD * 25 * 25 * 4);
  float*    fhat   = (float*)   alloc((size_t)BN * CD * 25 * 25 * 4);
  _Float16* z16    = (_Float16*)alloc((size_t)BN * CD * 25 * 25 * 2);
  float*    fd32   = (float*)   alloc((size_t)BN * CD * 25 * 25 * 4);
  _Float16* fd16   = (_Float16*)alloc((size_t)BN * CD * 25 * 25 * 2);
  float*    d0_32  = (float*)   alloc((size_t)BN * CD * 25 * 25 * 4);
  float*    g1     = (float*)   alloc((size_t)BN * CD * 50 * 50 * 4);
  float*    g2     = (float*)   alloc((size_t)BN * CD * 100 * 100 * 4);
  float*    g3     = (float*)   alloc((size_t)BN * C0 * YD * XD * 4);
  float*    cn     = (float*)   alloc(KC * 4);
  int*      counts = (int*)     alloc(KC * 4);
  int*      idxbuf = (int*)     alloc(2752 * 4);
  _Float16* wpk_e1 = (_Float16*)alloc((size_t)1024 * 128 * 2);
  _Float16* wpk_e2 = (_Float16*)alloc((size_t)2048 * 128 * 2);
  _Float16* wpk_e3 = (_Float16*)alloc((size_t)2048 * 128 * 2);
  _Float16* wpk_e4 = (_Float16*)alloc((size_t)1152 * 128 * 2);
  _Float16* wpk_pe[6];
  _Float16* wpk_pd[6];
  for (int i = 0; i < 6; ++i) wpk_pe[i] = (_Float16*)alloc((size_t)1152 * 128 * 2);
  for (int i = 0; i < 6; ++i) wpk_pd[i] = (_Float16*)alloc((size_t)1152 * 128 * 2);
  _Float16* wpk_d0 = (_Float16*)alloc((size_t)1152 * 128 * 2);

  const int SC[6]  = {1, 5, 10, 15, 20, 25};
  const int IOF[6] = {0, 2, 52, 252, 702, 1502};   // per-scale idx offsets (BN*s*s)

  auto conv = [&](const _Float16* in, const _Float16* wpk, const float* bias,
                  float* o32, _Float16* o16, int Cin, int H, int W, int Cout,
                  int Ho, int Wo, int kh, int kw, int st, int pd, int relu, int accum) {
    int M = BN * Ho * Wo;
    dim3 g((M + 15) / 16, 1);                 // 4 waves x 32N = 128 = Cout
    conv_wmma_kernel<<<g, 128, 0, stream>>>(in, wpk, bias, o32, o16, Cin, H, W,
                                            Cout, Ho, Wo, kh, kw, st, pd, relu, accum);
  };
  auto pack = [&](const float* w, _Float16* dst, int Cout, int Cin, int khw) {
    int n = Cout * Cin * khw;
    pack_w_kernel<<<(n + 255) / 256, 256, 0, stream>>>(w, dst, Cout, Cin, khw);
  };

  // scalar output slots: recon, vq_loss, 6 perps
  zero_f_kernel<<<1, 64, 0, stream>>>(out + NLOGITS, 8);

  // ---- encoder ----
  comp_onehot_kernel<<<BN * YD * XD, 64, 0, stream>>>(x, comp_w, h0_32);
  bn_stats_kernel<<<C0, 256, 0, stream>>>(h0_32, bn_m, bn_r, YD * XD);
  {
    int n = BN * C0 * YD * XD;
    bn_relu_f16_kernel<<<(n + 255) / 256, 256, 0, stream>>>(h0_32, bn_m, bn_r, bn_g,
                                                            bn_bb, h0_16, YD * XD, n);
  }
  pack(enc_w1, wpk_e1, 128, 64, 16);
  pack(enc_w2, wpk_e2, 128, 128, 16);
  pack(enc_w3, wpk_e3, 128, 128, 16);
  pack(enc_w4, wpk_e4, 128, 128, 9);
  for (int i = 0; i < 6; ++i) pack(phi_e_w + (size_t)i * 128 * 128 * 9, wpk_pe[i], 128, 128, 9);
  for (int i = 0; i < 6; ++i) pack(phi_d_w + (size_t)i * 128 * 128 * 9, wpk_pd[i], 128, 128, 9);
  pack(dec_w0, wpk_d0, 128, 128, 9);

  conv(h0_16, wpk_e1, enc_b1, nullptr, e1_16, 64, 200, 200, 128, 100, 100, 4, 4, 2, 1, 1, 0);
  conv(e1_16, wpk_e2, enc_b2, nullptr, e2_16, 128, 100, 100, 128, 50, 50, 4, 4, 2, 1, 1, 0);
  conv(e2_16, wpk_e3, enc_b3, nullptr, e3_16, 128, 50, 50, 128, 25, 25, 4, 4, 2, 1, 1, 0);
  conv(e3_16, wpk_e4, enc_b4, f32, nullptr, 128, 25, 25, 128, 25, 25, 3, 3, 1, 1, 0, 0);

  cnorm_kernel<<<(KC + 255) / 256, 256, 0, stream>>>(cbk, cn);

  // ---- multi-scale VQ (encoder residual loop) ----
  for (int i = 0; i < 6; ++i) {
    int s = SC[i];
    int* idx_s = idxbuf + IOF[i];
    {
      int n = BN * CD * s * s;
      area_kernel<<<(n + 255) / 256, 256, 0, stream>>>(f32, fhat, s);
    }
    zero_i_kernel<<<(KC + 255) / 256, 256, 0, stream>>>(counts, KC);
    float loss_scale = 1.25f / (float)(BN * CD * s * s);
    vq_kernel<<<BN * s * s, 256, 0, stream>>>(fhat, cbk, cn, idx_s, counts,
                                              out + NLOGITS + 1, s, loss_scale);
    perp_kernel<<<1, 256, 0, stream>>>(counts, out + NLOGITS + 2 + i,
                                       1.f / (float)(BN * s * s));
    {
      int n = BN * CD * HLAT * HLAT;
      bicubic_z_kernel<<<(n + 255) / 256, 256, 0, stream>>>(idx_s, cbk, z16, s);
    }
    conv(z16, wpk_pe[i], phi_e_b + (size_t)i * CD, f32, nullptr,
         128, 25, 25, 128, 25, 25, 3, 3, 1, 1, 1, -1);     // f -= relu(phi_enc(z))
  }

  // ---- decoder accumulation ----
  {
    int n = BN * CD * 25 * 25;
    zero_f_kernel<<<(n + 255) / 256, 256, 0, stream>>>(fd32, n);
  }
  for (int i = 0; i < 6; ++i) {
    int s = SC[i];
    int* idx_s = idxbuf + IOF[i];
    int n = BN * CD * HLAT * HLAT;
    bicubic_z_kernel<<<(n + 255) / 256, 256, 0, stream>>>(idx_s, cbk, z16, s);
    conv(z16, wpk_pd[i], phi_d_b + (size_t)i * CD, fd32, nullptr,
         128, 25, 25, 128, 25, 25, 3, 3, 1, 1, 1, 1);      // fd += relu(phi_dec(z))
  }
  {
    int n = BN * CD * 25 * 25;
    f32_to_f16_kernel<<<(n + 255) / 256, 256, 0, stream>>>(fd32, fd16, n);
  }
  conv(fd16, wpk_d0, dec_b0, d0_32, nullptr, 128, 25, 25, 128, 25, 25, 3, 3, 1, 1, 1, 0);

  {
    size_t n1 = (size_t)BN * CD * 50 * 50;
    convT_kernel<<<(unsigned)((n1 + 255) / 256), 256, 0, stream>>>(
        d0_32, dt_w1, dt_b1, g1, 128, 25, 25, 128, 50, 50, 1);
    size_t n2 = (size_t)BN * CD * 100 * 100;
    convT_kernel<<<(unsigned)((n2 + 255) / 256), 256, 0, stream>>>(
        g1, dt_w2, dt_b2, g2, 128, 50, 50, 128, 100, 100, 1);
    size_t n3 = (size_t)BN * C0 * YD * XD;
    convT_kernel<<<(unsigned)((n3 + 255) / 256), 256, 0, stream>>>(
        g2, dt_w3, dt_b3, g3, 128, 100, 100, 64, 200, 200, 0);
  }
  decomp_kernel<<<(NLOGITS + 255) / 256, 256, 0, stream>>>(g3, dcmp_w, dcmp_b, out);
  bce_kernel<<<4096, 256, 0, stream>>>(out, x, out + NLOGITS);
}
